// SelectiveStateSpace_45165876085134
// MI455X (gfx1250) — compile-verified, matrix operand order checked by reading, not run on hardware
//
#include <hip/hip_runtime.h>
#include <hip/hip_bf16.h>
#include <math.h>

// Problem dims (fixed by reference)
#define D_MODEL 1024
#define SEQ     2048
#define BATCH   8
#define CHUNK_T 16
#define NCHUNK  (SEQ / CHUNK_T)      // 128
#define R_DIM   (NCHUNK * BATCH)     // 1024 recurrence rows per step
#define M_TOT   (SEQ * BATCH)        // 16384 total rows

typedef __bf16 bf16;
typedef __attribute__((ext_vector_type(16))) __bf16 v16bf;
typedef __attribute__((ext_vector_type(8)))  __bf16 v8bf;
typedef __attribute__((ext_vector_type(4)))  __bf16 v4bf;
typedef __attribute__((ext_vector_type(8)))  float  v8f;
typedef __attribute__((ext_vector_type(4)))  float  v4f;

// GEMM tiling
#define BM 128
#define BN 128
#define BK 32
#define LDSS 40   // LDS row stride in bf16 elems (80B; multiple of 16B)

// ---- CDNA5 async global->LDS path (GLOBAL_LOAD_ASYNC_TO_LDS_B128) ----------
// Inline asm (bridge-doc recommended portable form). vdst = LDS byte address
// (flat->LDS mapping is addr[31:0], ISA 10.2); global address in a VGPR pair.
__device__ __forceinline__ void async_copy_b128(const void* gsrc, void* ldst) {
  unsigned ldsaddr = (unsigned)(unsigned long long)ldst;
  asm volatile("global_load_async_to_lds_b128 %0, %1, off"
               :
               : "v"(ldsaddr), "v"(gsrc)
               : "memory");
}

__device__ __forceinline__ void wait_async_lds() {
#if defined(__has_builtin)
#  if __has_builtin(__builtin_amdgcn_s_wait_asynccnt)
  __builtin_amdgcn_s_wait_asynccnt(0);
#  else
  asm volatile("s_wait_asynccnt 0x0" ::: "memory");
#  endif
#else
  asm volatile("s_wait_asynccnt 0x0" ::: "memory");
#endif
}

// ---------------------------------------------------------------------------
// Fragment load from LDS tile (row-major [rows][BK] with stride LDSS).
// 16-bit A-matrix layout (ISA 7.12.2): lane L holds row (L&15); half-wave
// selects K-groups: VGPR0-3 hold K = hi*8 .. hi*8+7, VGPR4-7 hold +16.
// ---------------------------------------------------------------------------
__device__ __forceinline__ v16bf load_frag(const bf16* base, int lane) {
  int i  = lane & 15;
  int hi = lane >> 4;
  const bf16* row = base + i * LDSS;
  v8bf lo = *(const v8bf*)(row + hi * 8);
  v8bf hv = *(const v8bf*)(row + 16 + hi * 8);
  v16bf f;
#pragma unroll
  for (int j = 0; j < 8; ++j) { f[j] = lo[j]; f[8 + j] = hv[j]; }
  return f;
}

// Stage a 128x32 f32 tile (convert to bf16) into LDS via VGPRs. 256 threads.
__device__ __forceinline__ void stage_f32(const float* __restrict__ src, int rowBase,
                                          int k0, bf16* lds, int tid) {
  int r = tid >> 1;
  int h = (tid & 1) * 16;
  const float* p = src + (size_t)(rowBase + r) * D_MODEL + k0 + h;
  v4f f0 = *(const v4f*)(p);
  v4f f1 = *(const v4f*)(p + 4);
  v4f f2 = *(const v4f*)(p + 8);
  v4f f3 = *(const v4f*)(p + 12);
  v8bf o0, o1;
#pragma unroll
  for (int j = 0; j < 4; ++j) {
    o0[j]     = (bf16)f0[j];  o0[4 + j] = (bf16)f1[j];
    o1[j]     = (bf16)f2[j];  o1[4 + j] = (bf16)f3[j];
  }
  bf16* q = lds + r * LDSS + h;
  *(v8bf*)(q)     = o0;
  *(v8bf*)(q + 8) = o1;
}

// Stage a 128x32 bf16 tile into LDS via async direct-to-LDS (no VGPR round trip).
__device__ __forceinline__ void stage_bf16(const bf16* __restrict__ src, int rowBase,
                                           int k0, bf16* lds, int tid) {
  int r = tid >> 1;
  int h = (tid & 1) * 16;
  const bf16* p = src + (size_t)(rowBase + r) * D_MODEL + k0 + h;
  bf16* q = lds + r * LDSS + h;
  async_copy_b128(p,     q);
  async_copy_b128(p + 8, q + 8);
}

// ---------------------------------------------------------------------------
// Unified WMMA GEMM: out[m,n] = epilogue( sum_k Asrc[m,k] * W[n,k] )
// MODE 0: Asrc = x (f32). out = XA (bf16), rows remapped to [t][c*8+b].
// MODE 1: Asrc = x (f32). out = sigmoid(acc + bias) -> sel (bf16), remapped.
// MODE 2: Asrc = H_{t-1} (bf16, 1024 rows). out = tanh(acc + XA_t)*sel_t -> H_t.
// MODE 3: Asrc = H (bf16, 16384 rows [t][r]). out = acc + x*D -> d_out (f32).
// ---------------------------------------------------------------------------
template <int MODE>
__global__ __launch_bounds__(256) void gemm_kernel(
    const float* __restrict__ aF32, const bf16* __restrict__ aBF,
    const bf16* __restrict__ W, bf16* __restrict__ outBF,
    float* __restrict__ outF32, const float* __restrict__ bias,
    const bf16* __restrict__ xaT, const bf16* __restrict__ selT,
    const float* __restrict__ xf32, const float* __restrict__ Dvec, int t) {
  __shared__ bf16 As[BM * LDSS];
  __shared__ bf16 Bs[BN * LDSS];

  int tid  = threadIdx.x;
  int lane = tid & 31;
  int wave = tid >> 5;
  int bm = blockIdx.y * BM;
  int bn = blockIdx.x * BN;
  int wm = (wave >> 2) * 64;   // wave tile: 64 (M) x 32 (N)
  int wn = (wave & 3) * 32;

  v8f acc[4][2];
#pragma unroll
  for (int i = 0; i < 4; ++i)
#pragma unroll
    for (int j = 0; j < 2; ++j) acc[i][j] = (v8f)0.0f;

  bool doK = !(MODE == 2 && t == 0);
  if (doK) {
    for (int k0 = 0; k0 < D_MODEL; k0 += BK) {
      if (MODE == 0 || MODE == 1) stage_f32(aF32, bm, k0, As, tid);
      else                        stage_bf16(aBF, bm, k0, As, tid);
      stage_bf16(W, bn, k0, Bs, tid);
      wait_async_lds();           // this wave's async LDS writes complete
      __syncthreads();            // + dscnt wait + block barrier
      v16bf af[4], bfr[2];
#pragma unroll
      for (int i = 0; i < 4; ++i) af[i] = load_frag(As + (wm + i * 16) * LDSS, lane);
#pragma unroll
      for (int j = 0; j < 2; ++j) bfr[j] = load_frag(Bs + (wn + j * 16) * LDSS, lane);
#pragma unroll
      for (int i = 0; i < 4; ++i)
#pragma unroll
        for (int j = 0; j < 2; ++j)
          acc[i][j] = __builtin_amdgcn_wmma_f32_16x16x32_bf16(
              false, af[i], false, bfr[j], (short)0, acc[i][j], false, false);
      __syncthreads();
    }
  }

  // Epilogue. C/D layout: lanes 0-15 hold M = v, lanes 16-31 hold M = 8+v;
  // N = lane&15. Each accumulator column spans an aligned 8-row group, so
  // every mode reduces to base + v*D_MODEL.
  int nloc  = lane & 15;
  int mhalf = (lane >> 4) * 8;
#pragma unroll
  for (int i = 0; i < 4; ++i) {
#pragma unroll
    for (int j = 0; j < 2; ++j) {
      int n = bn + wn + j * 16 + nloc;
      int mbase = bm + wm + i * 16 + mhalf;       // multiple of 8; b == v
      size_t base;
      if (MODE == 0 || MODE == 1) {
        int s0 = mbase >> 3;                      // chunk-time index, const over v
        int c = s0 >> 4, tt = s0 & 15;
        base = ((size_t)tt * R_DIM + c * 8) * D_MODEL + n;
      } else if (MODE == 2) {
        base = (size_t)mbase * D_MODEL + n;
      } else {                                    // MODE 3
        int tt = mbase >> 10;
        int c  = (mbase & 1023) >> 3;
        base = ((size_t)(c * 16 + tt) * BATCH) * D_MODEL + n;
      }
      float bval = (MODE == 1) ? bias[n] : 0.0f;
      float dval = (MODE == 3) ? Dvec[n] : 0.0f;
#pragma unroll
      for (int v = 0; v < 8; ++v) {
        size_t idx = base + (size_t)v * D_MODEL;
        float val = acc[i][j][v];
        if (MODE == 0) {
          outBF[idx] = (bf16)val;
        } else if (MODE == 1) {
          outBF[idx] = (bf16)(1.0f / (1.0f + __expf(-(val + bval))));
        } else if (MODE == 2) {
          float pre = val + (float)xaT[idx];
          outBF[idx] = (bf16)(tanhf(pre) * (float)selT[idx]);
        } else {  // MODE 3
          outF32[idx] = val + xf32[idx] * dval;
        }
      }
    }
  }
}

// f32 -> bf16 weight conversion
__global__ __launch_bounds__(256) void cvt_f32_to_bf16_kernel(
    const float* __restrict__ s, bf16* __restrict__ d, int n) {
  int i = (blockIdx.x * 256 + threadIdx.x) * 4;
  if (i + 3 < n) {
    v4f v = *(const v4f*)(s + i);
    v4bf o;
#pragma unroll
    for (int j = 0; j < 4; ++j) o[j] = (bf16)v[j];
    *(v4bf*)(d + i) = o;
  }
}

// One wave (32 lanes) per row of 1024; 8 rows per block. In-place on d_out.
__global__ __launch_bounds__(256) void layernorm_kernel(
    float* __restrict__ out, const float* __restrict__ gamma,
    const float* __restrict__ beta) {
  int wave = threadIdx.x >> 5;
  int lane = threadIdx.x & 31;
  size_t row = (size_t)blockIdx.x * 8 + wave;
  float* p = out + row * D_MODEL;

  v4f vals[8];
  float sum = 0.0f, sumsq = 0.0f;
#pragma unroll
  for (int j = 0; j < 8; ++j) {
    v4f v = *(const v4f*)(p + (j * 32 + lane) * 4);
    vals[j] = v;
#pragma unroll
    for (int u = 0; u < 4; ++u) { sum += v[u]; sumsq += v[u] * v[u]; }
  }
#pragma unroll
  for (int off = 16; off > 0; off >>= 1) {
    sum   += __shfl_xor(sum, off, 32);
    sumsq += __shfl_xor(sumsq, off, 32);
  }
  float mean = sum * (1.0f / D_MODEL);
  float var  = sumsq * (1.0f / D_MODEL) - mean * mean;
  float inv  = rsqrtf(var + 1e-5f);
#pragma unroll
  for (int j = 0; j < 8; ++j) {
    int c0 = (j * 32 + lane) * 4;
    v4f v = vals[j];
    v4f g = *(const v4f*)(gamma + c0);
    v4f b = *(const v4f*)(beta + c0);
#pragma unroll
    for (int u = 0; u < 4; ++u) v[u] = (v[u] - mean) * inv * g[u] + b[u];
    *(v4f*)(p + c0) = v;
  }
}

extern "C" void kernel_launch(void* const* d_in, const int* in_sizes, int n_in,
                              void* d_out, int out_size, void* d_ws, size_t ws_size,
                              hipStream_t stream) {
  const float* x     = (const float*)d_in[0];
  const float* Amat  = (const float*)d_in[1];
  const float* Bmat  = (const float*)d_in[2];
  const float* Cmat  = (const float*)d_in[3];
  const float* Dvec  = (const float*)d_in[4];
  const float* Wsel  = (const float*)d_in[5];
  const float* bsel  = (const float*)d_in[6];
  const float* gamma = (const float*)d_in[7];
  const float* beta  = (const float*)d_in[8];
  float* out = (float*)d_out;

  // Workspace layout (~104 MB): 4 bf16 weights (8MB), XA/sel/H bf16 (32MB each)
  char* ws = (char*)d_ws;
  const size_t WB = (size_t)D_MODEL * D_MODEL * sizeof(bf16);
  bf16* Abf = (bf16*)(ws);
  bf16* Bbf = (bf16*)(ws + WB);
  bf16* Cbf = (bf16*)(ws + 2 * WB);
  bf16* Sbf = (bf16*)(ws + 3 * WB);
  bf16* XA  = (bf16*)(ws + 4 * WB);
  bf16* SEL = XA + (size_t)M_TOT * D_MODEL;
  bf16* H   = SEL + (size_t)M_TOT * D_MODEL;

  const int nW = D_MODEL * D_MODEL;
  dim3 blk(256);
  cvt_f32_to_bf16_kernel<<<nW / 1024, blk, 0, stream>>>(Amat, Abf, nW);
  cvt_f32_to_bf16_kernel<<<nW / 1024, blk, 0, stream>>>(Bmat, Bbf, nW);
  cvt_f32_to_bf16_kernel<<<nW / 1024, blk, 0, stream>>>(Cmat, Cbf, nW);
  cvt_f32_to_bf16_kernel<<<nW / 1024, blk, 0, stream>>>(Wsel, Sbf, nW);

  dim3 gBig(D_MODEL / BN, M_TOT / BM);  // (8, 128)
  gemm_kernel<0><<<gBig, blk, 0, stream>>>(x, nullptr, Abf, XA, nullptr,
                                           nullptr, nullptr, nullptr, nullptr, nullptr, 0);
  gemm_kernel<1><<<gBig, blk, 0, stream>>>(x, nullptr, Sbf, SEL, nullptr,
                                           bsel, nullptr, nullptr, nullptr, nullptr, 0);

  // Sequential recurrence: H_t = tanh(H_{t-1} @ B^T + XA_t) * sel_t
  dim3 gRec(D_MODEL / BN, R_DIM / BM);  // (8, 8)
  for (int t = 0; t < CHUNK_T; ++t) {
    const bf16* Hprev = (t == 0) ? H : (H + (size_t)(t - 1) * R_DIM * D_MODEL);
    gemm_kernel<2><<<gRec, blk, 0, stream>>>(
        nullptr, Hprev, Bbf, H + (size_t)t * R_DIM * D_MODEL, nullptr, nullptr,
        XA + (size_t)t * R_DIM * D_MODEL, SEL + (size_t)t * R_DIM * D_MODEL,
        nullptr, nullptr, t);
  }

  // d_out = H @ C^T + x * D  (rows remapped back to sequence order)
  gemm_kernel<3><<<gBig, blk, 0, stream>>>(nullptr, H, Cbf, nullptr, out,
                                           nullptr, nullptr, nullptr, x, Dvec, 0);

  layernorm_kernel<<<M_TOT / 8, blk, 0, stream>>>(out, gamma, beta);
}
